// TemporalFeatureExtractor_18880676233651
// MI455X (gfx1250) — matrix operand order repaired and checked by reading, not compiled, not run on hardware
//
#include <hip/hip_runtime.h>

typedef __attribute__((ext_vector_type(16))) __bf16 v16bf;
typedef __attribute__((ext_vector_type(8)))  __bf16 v8bf;
typedef __attribute__((ext_vector_type(4)))  __bf16 v4bf;
typedef __attribute__((ext_vector_type(8)))  float  v8f;
typedef __attribute__((ext_vector_type(4)))  float  v4f;

#define S_LEN 2048
#define D_DIM 64
#define KW 15
#define KTOT (D_DIM * KW)        // 960
#define KSTEPS (KTOT / 32)       // 30
#define KK0_LO 8                 // tile0 (short+medium) K window: taps 4..10
#define KK0_HI 21                //   = Kglob [256,704) = ksteps 8..21
#define CHUNK 128
#define HALO_ROWS (CHUNK + 14)   // 142
#define LDS_ROWS 144
#define LDS_STRIDE 72            // padded (144 B) to dodge bank conflicts; 16B aligned

// ws layout (bytes)
#define WMAT_ELEMS (2 * KSTEPS * 32 * 16)   // 30720 bf16 (2 N-tiles)
#define BIAS_OFF   (WMAT_ELEMS * 2)         // 61440
#define ACC_OFF    (BIAS_OFF + 128)         // 61568 : float[256][32]

// ---------------------------------------------------------------------------
// Kernel 1: build combined bf16 weight matrix in B-fragment lane layout,
// pack combined bias, zero the per-batch channel accumulator.
// B-fragment, contiguous K order used here (32K x 16N): lane<16 -> N=lane, K=e (0..15);
// lane>=16 -> N=lane-16, K=16+e.  Stored [tile][kstep][lane][e] contiguous.
// ---------------------------------------------------------------------------
__global__ void tfe_prep_kernel(const float* __restrict__ w_s,
                                const float* __restrict__ b_s,
                                const float* __restrict__ w_m,
                                const float* __restrict__ b_m,
                                const float* __restrict__ w_l,
                                const float* __restrict__ b_l,
                                __bf16* __restrict__ wmat,
                                float* __restrict__ bias32,
                                float* __restrict__ acc) {
    int idx = blockIdx.x * blockDim.x + threadIdx.x;
    if (idx < WMAT_ELEMS) {
        int tile = idx / (KSTEPS * 32 * 16);
        int rem  = idx % (KSTEPS * 32 * 16);
        int kk   = rem / (32 * 16);
        int rem2 = rem % (32 * 16);
        int lane = rem2 >> 4;
        int e    = rem2 & 15;
        int half = lane >> 4;
        int n    = (lane & 15) + tile * 16;      // output channel (0..31)
        int kg   = kk * 32 + half * 16 + e;      // global K (0..959)
        int j    = kg >> 6;                      // tap in the 15-wide window
        int d    = kg & 63;                      // input channel
        float w = 0.f;
        if (n < 8) {                             // short: taps 6..8
            int js = j - 6;
            if (js >= 0 && js < 3) w = w_s[(n * D_DIM + d) * 3 + js];
        } else if (n < 16) {                     // medium: taps 4..10
            int jm = j - 4;
            if (jm >= 0 && jm < 7) w = w_m[((n - 8) * D_DIM + d) * 7 + jm];
        } else if (n < 24) {                     // long: taps 0..14
            w = w_l[((n - 16) * D_DIM + d) * KW + j];
        }
        wmat[idx] = (__bf16)w;
    }
    if (idx < 256 * 32) acc[idx] = 0.f;
    if (idx < 32) {
        float bv = 0.f;
        if (idx < 8)       bv = b_s[idx];
        else if (idx < 16) bv = b_m[idx - 8];
        else if (idx < 24) bv = b_l[idx - 16];
        bias32[idx] = bv;
    }
}

// ---------------------------------------------------------------------------
// Kernel 2: fused conv(3 branches) + bias + ReLU + time-sum.
// 1 workgroup = 1 batch x 128 time steps; 8 waves, each a 16-wide time tile.
// ---------------------------------------------------------------------------
__global__ __launch_bounds__(256) void tfe_conv_kernel(
    const float*  __restrict__ x,       // [256][2048][64] f32
    const __bf16* __restrict__ wmat,    // pre-swizzled B fragments
    const float*  __restrict__ bias32,
    float*        __restrict__ acc) {   // [256][32] f32 (summed over time)
    __shared__ __bf16 lx[LDS_ROWS * LDS_STRIDE];
    __shared__ float  csum[32];

    const int tid = threadIdx.x;
    const int b   = blockIdx.y;
    const int t0  = blockIdx.x * CHUNK;

    // Stage x chunk + 7-row halo into LDS as bf16.
    // float4 non-temporal loads: x is streamed exactly once -> NT keeps the
    // hot weight fragments resident in the read-only WGP$.
    for (int i = tid; i < (LDS_ROWS * D_DIM) / 4; i += 256) {
        int r = i >> 4;             // time row in LDS
        int d = (i & 15) << 2;      // input-channel quad
        int t = t0 - 7 + r;
        v4f xv = (v4f)(0.f);
        if (r < HALO_ROWS && t >= 0 && t < S_LEN)
            xv = __builtin_nontemporal_load(
                     (const v4f*)(x + ((b * S_LEN + t) * D_DIM + d)));
        v4bf cv;
        cv[0] = (__bf16)xv[0]; cv[1] = (__bf16)xv[1];
        cv[2] = (__bf16)xv[2]; cv[3] = (__bf16)xv[3];
        *(v4bf*)&lx[r * LDS_STRIDE + d] = cv;
    }
    if (tid < 32) csum[tid] = 0.f;
    __syncthreads();

    const int wave = tid >> 5;
    const int lane = tid & 31;
    const int half = lane >> 4;       // A/K-group select per 16-bit A layout
    const int m    = lane & 15;       // time row within tile
    const int ttl  = wave * 16;       // tile base (relative to t0-7 halo start)
    const int n0   = lane & 15;       // output channel (N) within tile

    // C-init = per-element bias (applied before ReLU, per time position).
    float bias0 = bias32[n0];
    float bias1 = bias32[16 + n0];
    v8f acc0, acc1;
#pragma unroll
    for (int r = 0; r < 8; ++r) { acc0[r] = bias0; acc1[r] = bias1; }

    const v16bf* wm = (const v16bf*)wmat;
#pragma unroll
    for (int kk = 0; kk < KSTEPS; ++kk) {
        // A fragment (16x32 bf16): half 0 holds K {0..15}, half 1 {16..31}
        int kb0 = kk * 32 + half * 16;
        int kb1 = kb0 + 8;
        int j0 = kb0 >> 6, d0 = kb0 & 63;   // 8 contiguous K stay in one tap row
        int j1 = kb1 >> 6, d1 = kb1 & 63;
        union { v16bf v; v8bf h[2]; } ua;
        ua.h[0] = *(const v8bf*)&lx[(ttl + m + j0) * LDS_STRIDE + d0];
        ua.h[1] = *(const v8bf*)&lx[(ttl + m + j1) * LDS_STRIDE + d1];

        // tile1 (long branch, channels 16..23): full K range
        v16bf bm1 = wm[(1 * KSTEPS + kk) * 32 + lane];
        acc1 = __builtin_amdgcn_wmma_f32_16x16x32_bf16(
            false, ua.v, false, bm1, (short)0, acc1, false, false);

        // tile0 (short+medium, channels 0..15): weights are identically zero
        // outside taps 4..10 -> only ksteps 8..21 contribute.
        if (kk >= KK0_LO && kk <= KK0_HI) {
            v16bf bm0 = wm[(0 * KSTEPS + kk) * 32 + lane];
            acc0 = __builtin_amdgcn_wmma_f32_16x16x32_bf16(
                false, ua.v, false, bm0, (short)0, acc0, false, false);
        }
    }

    // ReLU + sum over this tile's 16 time positions (8 per lane half).
    float s0 = 0.f, s1 = 0.f;
#pragma unroll
    for (int r = 0; r < 8; ++r) {
        s0 += fmaxf(acc0[r], 0.f);
        s1 += fmaxf(acc1[r], 0.f);
    }
    atomicAdd(&csum[n0], s0);           // ds_add_f32
    atomicAdd(&csum[16 + n0], s1);
    __syncthreads();
    if (tid < 24) atomicAdd(&acc[b * 32 + tid], csum[tid]);  // global_atomic_add_f32
}

// ---------------------------------------------------------------------------
// Kernel 3: mean + FC1(ReLU) + FC2.  One thread per batch row; tiny.
// ---------------------------------------------------------------------------
__global__ void tfe_head_kernel(const float* __restrict__ acc,
                                const float* __restrict__ w_fc1,
                                const float* __restrict__ b_fc1,
                                const float* __restrict__ w_fc2,
                                const float* __restrict__ b_fc2,
                                float* __restrict__ out) {
    int b = threadIdx.x;
    if (b >= 256) return;
    float c[24];
#pragma unroll
    for (int i = 0; i < 24; ++i) c[i] = acc[b * 32 + i] * (1.f / 2048.f);
    float h[32];
#pragma unroll
    for (int j = 0; j < 32; ++j) {
        float s = b_fc1[j];
#pragma unroll
        for (int i = 0; i < 24; ++i) s += c[i] * w_fc1[j * 24 + i];
        h[j] = fmaxf(s, 0.f);
    }
#pragma unroll
    for (int j2 = 0; j2 < 32; ++j2) {
        float s = b_fc2[j2];
#pragma unroll
        for (int j = 0; j < 32; ++j) s += h[j] * w_fc2[j2 * 32 + j];
        out[b * 32 + j2] = s;
    }
}

extern "C" void kernel_launch(void* const* d_in, const int* in_sizes, int n_in,
                              void* d_out, int out_size, void* d_ws, size_t ws_size,
                              hipStream_t stream) {
    const float* x      = (const float*)d_in[0];
    const float* w_s    = (const float*)d_in[1];
    const float* b_s    = (const float*)d_in[2];
    const float* w_m    = (const float*)d_in[3];
    const float* b_m    = (const float*)d_in[4];
    const float* w_l    = (const float*)d_in[5];
    const float* b_l    = (const float*)d_in[6];
    const float* w_fc1  = (const float*)d_in[7];
    const float* b_fc1  = (const float*)d_in[8];
    const float* w_fc2  = (const float*)d_in[9];
    const float* b_fc2  = (const float*)d_in[10];

    char*   ws     = (char*)d_ws;
    __bf16* wmat   = (__bf16*)(ws);
    float*  bias32 = (float*)(ws + BIAS_OFF);
    float*  acc    = (float*)(ws + ACC_OFF);

    tfe_prep_kernel<<<(WMAT_ELEMS + 255) / 256, 256, 0, stream>>>(
        w_s, b_s, w_m, b_m, w_l, b_l, wmat, bias32, acc);

    dim3 grid(S_LEN / CHUNK, 256);
    tfe_conv_kernel<<<grid, 256, 0, stream>>>(x, wmat, bias32, acc);

    tfe_head_kernel<<<1, 256, 0, stream>>>(acc, w_fc1, b_fc1, w_fc2, b_fc2,
                                           (float*)d_out);
}
